// SSD_MobileNetV2_HEAD_9929964388752
// MI455X (gfx1250) — matrix operand order repaired and checked by reading, not compiled, not run on hardware
//
#include <hip/hip_runtime.h>
#include <hip/hip_bf16.h>

typedef __attribute__((ext_vector_type(16))) __bf16 v16bf;
typedef __attribute__((ext_vector_type(8)))  float  v8f;

// Explicit global-address-space pointers: keep fragment loads on the
// GLOBAL_LOAD path (LOADcnt only) instead of FLAT (LOADcnt+DScnt).
typedef __bf16 __attribute__((address_space(1))) gbf16_t;
typedef v16bf  __attribute__((address_space(1))) gv16bf_t;

#define BN_EPS 1e-5f
#define NPRIORS 2268
#define BATCH 64

// Zero page: .bss device global, zero-initialized at module load, never written.
__device__ __attribute__((aligned(256))) float g_zero[5776];

// ============================================================================
// Packed path: operands pre-packed to bf16 fragment layouts (contiguous k = 16h + s).
// ============================================================================

// NCHW f32 -> [b][y][x][cin] bf16 (channels-last)
template<int F, int CIN>
__global__ __launch_bounds__(256)
void pack_src(const float* __restrict__ src, __bf16* __restrict__ dst)
{
    constexpr int F2 = F * F;
    const size_t o = (size_t)blockIdx.x * 256 + threadIdx.x;
    if (o >= (size_t)BATCH * F2 * CIN) return;
    const int c  = (int)(o % CIN);
    const int n  = (int)(o / CIN);          // b*F2 + yx
    const int b  = n / F2;
    const int yx = n % F2;
    dst[o] = (__bf16)src[((size_t)b * CIN + c) * F2 + yx];
}

// OIHW f32 -> A-fragment order: dst[((m*9+r)*CIN) + c32*32 + h*16 + j]
//   = w[(m*CIN + (c32*32 + h*16 + j))*9 + r]
template<int CIN>
__global__ __launch_bounds__(256)
void pack_w(const float* __restrict__ w, __bf16* __restrict__ dst, int Cout)
{
    const size_t o = (size_t)blockIdx.x * 256 + threadIdx.x;
    if (o >= (size_t)Cout * 9 * CIN) return;
    const int m   = (int)(o / (9 * CIN));
    const int t1  = (int)(o % (9 * CIN));
    const int r   = t1 / CIN;
    const int t2  = t1 % CIN;
    const int c32 = t2 / 32;
    const int t3  = t2 % 32;
    const int h   = t3 / 16;
    const int j   = t3 % 16;
    const int cin = c32 * 32 + h * 16 + j;
    dst[o] = (__bf16)w[((size_t)m * CIN + cin) * 9 + r];
}

// Implicit-GEMM conv3x3(pad=1)+BN on pre-packed bf16 operands.
// One wave computes a 16(M) x 64(N) tile; fragments load as 2x global b128.
template<int F, int CIN, int VALS>
__global__ __launch_bounds__(256)
void conv_head_wmma_pk(const __bf16* __restrict__ sp,  // packed src
                       const __bf16* __restrict__ ap,  // packed weights
                       const float* __restrict__ gamma,
                       const float* __restrict__ beta,
                       const float* __restrict__ mean,
                       const float* __restrict__ var,
                       float* __restrict__ out,
                       int Cout, int nAnchors, int priorBase)
{
    constexpr int F2   = F * F;
    constexpr int NCOL = BATCH * F2;
    constexpr int NTB  = NCOL / 64;

    const int lane = threadIdx.x & 31;
    const int wave = threadIdx.x >> 5;
    const int tilesM = (Cout + 15) >> 4;

    const int tile = blockIdx.x * 8 + wave;
    if (tile >= tilesM * NTB) return;            // wave-uniform exit

    const int mt  = tile / NTB;
    const int ntb = tile % NTB;
    const int h   = lane >> 4;
    const int ml  = lane & 15;

    const gbf16_t* zpage = (const gbf16_t*)(const __bf16*)(const void*)g_zero;

    const int  m    = mt * 16 + ml;
    const bool mok  = (m < Cout);
    const int  aadv = mok ? 32 : 0;

    int bb[4], rem[4], y0[4], x0[4];
#pragma unroll
    for (int nb = 0; nb < 4; ++nb) {
        const int n = ntb * 64 + nb * 16 + ml;
        bb[nb]  = n / F2;
        rem[nb] = n % F2;
        y0[nb]  = rem[nb] / F;
        x0[nb]  = rem[nb] % F;
    }

    v8f acc[4] = {v8f{}, v8f{}, v8f{}, v8f{}};

    for (int r = 0; r < 9; ++r) {
        const int ky = r / 3 - 1;
        const int kx = r % 3 - 1;

        const gbf16_t* bptr[4];
        int badv[4];
        int any = 0;
#pragma unroll
        for (int nb = 0; nb < 4; ++nb) {
            const int yv = y0[nb] + ky;
            const int xv = x0[nb] + kx;
            const bool inb = (yv >= 0) && (yv < F) && (xv >= 0) && (xv < F);
            any |= (int)inb;
            const gbf16_t* real = (const gbf16_t*)
                (sp + (size_t)(bb[nb] * F2 + yv * F + xv) * CIN + h * 16);
            bptr[nb] = inb ? real : zpage;
            badv[nb] = inb ? 32 : 0;
        }
        if (!__any(any)) continue;               // wave-uniform tap skip

        const gbf16_t* aptr = mok
            ? (const gbf16_t*)(ap + ((size_t)m * 9 + r) * CIN + h * 16)
            : zpage;

#pragma unroll 2
        for (int c0 = 0; c0 < CIN; c0 += 32) {
            const v16bf afrag = *(const gv16bf_t*)aptr;    // 2x global b128
#pragma unroll
            for (int nb = 0; nb < 4; ++nb) {
                const v16bf bfrag = *(const gv16bf_t*)bptr[nb];
                acc[nb] = __builtin_amdgcn_wmma_f32_16x16x32_bf16(
                              false, afrag, false, bfrag, (short)0,
                              acc[nb], false, false);
            }
            aptr += aadv;
#pragma unroll
            for (int nb = 0; nb < 4; ++nb) bptr[nb] += badv[nb];
        }
    }

#pragma unroll
    for (int g = 0; g < 8; ++g) {
        const int mm = mt * 16 + h * 8 + g;
        if (mm < Cout) {
            const float s = gamma[mm] * rsqrtf(var[mm] + BN_EPS);
            const float t = beta[mm] - mean[mm] * s;
            const int a_idx = mm / VALS;
            const int sub   = mm % VALS;
#pragma unroll
            for (int nb = 0; nb < 4; ++nb) {
                const int prior = priorBase + rem[nb] * nAnchors + a_idx;
                out[((size_t)bb[nb] * NPRIORS + prior) * VALS + sub] =
                    acc[nb][g] * s + t;
            }
        }
    }
}

// ============================================================================
// Direct path (fallback when workspace is too small): loads f32 + converts.
// ============================================================================
template<int F, int CIN, int VALS>
__global__ __launch_bounds__(256)
void conv_head_wmma(const float* __restrict__ src,
                    const float* __restrict__ w,
                    const float* __restrict__ gamma,
                    const float* __restrict__ beta,
                    const float* __restrict__ mean,
                    const float* __restrict__ var,
                    float* __restrict__ out,
                    int Cout, int nAnchors, int priorBase)
{
    constexpr int F2   = F * F;
    constexpr int K    = CIN * 9;
    constexpr int NCOL = BATCH * F2;
    constexpr int NTB  = NCOL / 64;

    const int lane = threadIdx.x & 31;
    const int wave = threadIdx.x >> 5;
    const int tilesM = (Cout + 15) >> 4;

    const int tile = blockIdx.x * 8 + wave;
    if (tile >= tilesM * NTB) return;

    const int mt  = tile / NTB;
    const int ntb = tile % NTB;
    const int h   = lane >> 4;
    const int ml  = lane & 15;

    const int  m    = mt * 16 + ml;
    const bool mok  = (m < Cout);
    const int  aadv = mok ? 32 * 9 : 0;

    int bb[4], rem[4], y0[4], x0[4];
#pragma unroll
    for (int nb = 0; nb < 4; ++nb) {
        const int n = ntb * 64 + nb * 16 + ml;
        bb[nb]  = n / F2;
        rem[nb] = n % F2;
        y0[nb]  = rem[nb] / F;
        x0[nb]  = rem[nb] % F;
    }

    v8f acc[4] = {v8f{}, v8f{}, v8f{}, v8f{}};

    for (int r = 0; r < 9; ++r) {
        const int ky = r / 3 - 1;
        const int kx = r % 3 - 1;

        const float* bptr[4];
        int badv[4];
        int any = 0;
#pragma unroll
        for (int nb = 0; nb < 4; ++nb) {
            const int yv = y0[nb] + ky;
            const int xv = x0[nb] + kx;
            const bool inb = (yv >= 0) && (yv < F) && (xv >= 0) && (xv < F);
            any |= (int)inb;
            const float* real = src + (size_t)(bb[nb] * CIN + h * 16) * F2
                                    + (size_t)yv * F + xv;
            bptr[nb] = inb ? real : &g_zero[0];
            badv[nb] = inb ? 32 * F2 : 0;
        }
        if (!__any(any)) continue;

        const float* aptr = mok ? (w + (size_t)m * K + (size_t)(h * 16) * 9 + r)
                                : &g_zero[0];

        for (int c0 = 0; c0 < CIN; c0 += 32) {
            v16bf afrag;
#pragma unroll
            for (int j = 0; j < 16; ++j) {
                const int co = j;
                afrag[j] = (__bf16)aptr[co * 9];
            }
#pragma unroll
            for (int nb = 0; nb < 4; ++nb) {
                v16bf bfrag;
#pragma unroll
                for (int j = 0; j < 16; ++j)
                    bfrag[j] = (__bf16)bptr[nb][j * F2];
                acc[nb] = __builtin_amdgcn_wmma_f32_16x16x32_bf16(
                              false, afrag, false, bfrag, (short)0,
                              acc[nb], false, false);
            }
            aptr += aadv;
#pragma unroll
            for (int nb = 0; nb < 4; ++nb) bptr[nb] += badv[nb];
        }
    }

#pragma unroll
    for (int g = 0; g < 8; ++g) {
        const int mm = mt * 16 + h * 8 + g;
        if (mm < Cout) {
            const float s = gamma[mm] * rsqrtf(var[mm] + BN_EPS);
            const float t = beta[mm] - mean[mm] * s;
            const int a_idx = mm / VALS;
            const int sub   = mm % VALS;
#pragma unroll
            for (int nb = 0; nb < 4; ++nb) {
                const int prior = priorBase + rem[nb] * nAnchors + a_idx;
                out[((size_t)bb[nb] * NPRIORS + prior) * VALS + sub] =
                    acc[nb][g] * s + t;
            }
        }
    }
}

__global__ void copy_priors(const float* __restrict__ p, float* __restrict__ o, int n)
{
    const int i = blockIdx.x * 256 + threadIdx.x;
    if (i < n) o[i] = p[i];
}

// ---------------- host side ----------------

static const int FM_[6]    = {19, 10, 5, 3, 2, 1};
static const int CH_[6]    = {576, 1280, 512, 256, 256, 128};
static const int CFG_[6]   = {4, 6, 6, 6, 4, 4};
static const int PBASE_[6] = {0, 1444, 2044, 2194, 2248, 2264};

extern "C" void kernel_launch(void* const* d_in, const int* in_sizes, int n_in,
                              void* d_out, int out_size, void* d_ws, size_t ws_size,
                              hipStream_t stream)
{
    (void)in_sizes; (void)n_in; (void)out_size;

    const size_t LOC_TOTAL  = (size_t)BATCH * NPRIORS * 4;
    const size_t CONF_TOTAL = (size_t)BATCH * NPRIORS * 21;
    float* out_loc   = (float*)d_out;
    float* out_conf  = (float*)d_out + LOC_TOTAL;
    float* out_prior = (float*)d_out + LOC_TOTAL + CONF_TOTAL;

    // workspace layout (bf16 elements, 128-elem aligned)
    size_t soff[6], lwoff[6], cwoff[6], off = 0;
    for (int i = 0; i < 6; ++i) {
        soff[i] = off;
        off = (off + (size_t)BATCH * FM_[i] * FM_[i] * CH_[i] + 127) & ~(size_t)127;
    }
    for (int i = 0; i < 6; ++i) {
        lwoff[i] = off;
        off = (off + (size_t)(CFG_[i] * 4) * 9 * CH_[i] + 127) & ~(size_t)127;
    }
    for (int i = 0; i < 6; ++i) {
        cwoff[i] = off;
        off = (off + (size_t)(CFG_[i] * 21) * 9 * CH_[i] + 127) & ~(size_t)127;
    }
    const bool packed = (d_ws != nullptr) && (ws_size >= off * sizeof(__bf16));
    __bf16* wsb = (__bf16*)d_ws;

#define GRID_FOR(Cout, f) (((((Cout) + 15) >> 4) * ((BATCH * (f) * (f)) / 64) + 7) / 8)

#define DO_SRC(IDX, FF, CC)                                                      \
    case IDX: {                                                                  \
        const size_t ns = (size_t)BATCH * FF * FF * CC;                          \
        pack_src<FF, CC><<<(int)((ns + 255) / 256), 256, 0, stream>>>(           \
            (const float*)d_in[IDX], wsb + soff[IDX]);                           \
        const size_t nl = (size_t)(CFG_[IDX] * 4) * 9 * CC;                      \
        pack_w<CC><<<(int)((nl + 255) / 256), 256, 0, stream>>>(                 \
            (const float*)d_in[6 + IDX * 5], wsb + lwoff[IDX], CFG_[IDX] * 4);   \
        const size_t nc = (size_t)(CFG_[IDX] * 21) * 9 * CC;                     \
        pack_w<CC><<<(int)((nc + 255) / 256), 256, 0, stream>>>(                 \
            (const float*)d_in[36 + IDX * 5], wsb + cwoff[IDX], CFG_[IDX] * 21); \
    } break;

#define DO_GEMM(IDX, FF, CC)                                                     \
    case IDX: {                                                                  \
        const int cl = CFG_[IDX] * 4, cc = CFG_[IDX] * 21;                       \
        const float* s0 = (const float*)d_in[IDX];                               \
        const float* lw = (const float*)d_in[6 + IDX * 5];                       \
        const float* cw = (const float*)d_in[36 + IDX * 5];                      \
        const float *lg = (const float*)d_in[6 + IDX * 5 + 1],                   \
                    *lb = (const float*)d_in[6 + IDX * 5 + 2],                   \
                    *lm = (const float*)d_in[6 + IDX * 5 + 3],                   \
                    *lv = (const float*)d_in[6 + IDX * 5 + 4];                   \
        const float *cg = (const float*)d_in[36 + IDX * 5 + 1],                  \
                    *cb = (const float*)d_in[36 + IDX * 5 + 2],                  \
                    *cm = (const float*)d_in[36 + IDX * 5 + 3],                  \
                    *cv = (const float*)d_in[36 + IDX * 5 + 4];                  \
        if (packed) {                                                            \
            conv_head_wmma_pk<FF, CC, 4><<<GRID_FOR(cl, FF), 256, 0, stream>>>(  \
                wsb + soff[IDX], wsb + lwoff[IDX], lg, lb, lm, lv,               \
                out_loc, cl, CFG_[IDX], PBASE_[IDX]);                            \
            conv_head_wmma_pk<FF, CC, 21><<<GRID_FOR(cc, FF), 256, 0, stream>>>( \
                wsb + soff[IDX], wsb + cwoff[IDX], cg, cb, cm, cv,               \
                out_conf, cc, CFG_[IDX], PBASE_[IDX]);                           \
        } else {                                                                 \
            conv_head_wmma<FF, CC, 4><<<GRID_FOR(cl, FF), 256, 0, stream>>>(     \
                s0, lw, lg, lb, lm, lv, out_loc, cl, CFG_[IDX], PBASE_[IDX]);    \
            conv_head_wmma<FF, CC, 21><<<GRID_FOR(cc, FF), 256, 0, stream>>>(    \
                s0, cw, cg, cb, cm, cv, out_conf, cc, CFG_[IDX], PBASE_[IDX]);   \
        }                                                                        \
    } break;

    if (packed) {
        for (int i = 0; i < 6; ++i) {
            switch (i) {
                DO_SRC(0, 19, 576) DO_SRC(1, 10, 1280) DO_SRC(2, 5, 512)
                DO_SRC(3, 3, 256)  DO_SRC(4, 2, 256)   DO_SRC(5, 1, 128)
            }
        }
    }
    for (int i = 0; i < 6; ++i) {
        switch (i) {
            DO_GEMM(0, 19, 576) DO_GEMM(1, 10, 1280) DO_GEMM(2, 5, 512)
            DO_GEMM(3, 3, 256)  DO_GEMM(4, 2, 256)   DO_GEMM(5, 1, 128)
        }
    }
#undef DO_SRC
#undef DO_GEMM
#undef GRID_FOR

    const int np = NPRIORS * 4;
    copy_priors<<<(np + 255) / 256, 256, 0, stream>>>(
        (const float*)d_in[66], out_prior, np);
}
